// LSTM_46840913330760
// MI455X (gfx1250) — matrix operand order repaired and checked by reading, not compiled, not run on hardware
//
#include <hip/hip_runtime.h>
#include <math.h>

typedef __attribute__((ext_vector_type(16))) _Float16 v16h;
typedef __attribute__((ext_vector_type(8)))  _Float16 v8h;
typedef __attribute__((ext_vector_type(8)))  float    v8f;

namespace {

constexpr int Bsz = 512, Tt = 1024, IN_ = 44, H_ = 30;
constexpr int GP   = 128;   // padded gate columns: 4 gates x 32
constexpr int KW0  = 96;    // layer0 concat K: [x(44)pad64 | h(30)pad32]
constexpr int KW12 = 64;    // layers1/2:       [h_in pad32 | h  pad32]
constexpr int BT   = 16;    // batch tile per block (WMMA M)
constexpr int NTHREADS = 128; // 4 wave32s

// ---- dynamic LDS layout (bytes) ----
constexpr int SM_W0   = 0;                       // 128 x 96 f16  (B as [n][k])
constexpr int SM_W1   = SM_W0   + 128*KW0*2;     // 128 x 64 f16
constexpr int SM_W2   = SM_W1   + 128*KW12*2;    // 128 x 64 f16
constexpr int SM_WO1  = SM_W2   + 128*KW12*2;    // 32 x 32 f16
constexpr int SM_WO2  = SM_WO1  + 32*32*2;       // 32 x 32 f16
constexpr int SM_B0   = SM_WO2  + 32*32*2;       // 128 f32 (bih+bhh, padded)
constexpr int SM_B1   = SM_B0   + GP*4;
constexpr int SM_B2   = SM_B1   + GP*4;
constexpr int SM_BO1  = SM_B2   + GP*4;          // 32 f32
constexpr int SM_BO2  = SM_BO1  + 32*4;          // 32 f32
constexpr int SM_AX   = SM_BO2  + 32*4;          // 16 x 64 f16 (x_t staging, zero-padded)
constexpr int SM_HST  = SM_AX   + BT*64*2;       // 3 x 16 x 32 f16 (h per layer)
constexpr int SM_CST  = SM_HST  + 3*512*2;       // 3 x 16 x 32 f32 (c per layer)
constexpr int SM_GBUF = SM_CST  + 3*512*4;       // 16 x 128 f32 (activated gates)
constexpr int SM_LAST = SM_GBUF + BT*GP*4;       // 16 x 32 f16 relu(h2)
constexpr int SM_OBUF = SM_LAST + 512*2;         // 16 x 32 f16 relu(o1)
constexpr int SM_O2   = SM_OBUF + 512*2;         // 16 x 32 f32 o2
constexpr int SM_TOTAL= SM_O2   + BT*32*4;       // ~86.8 KB  (<320 KB WGP LDS)

struct ASrc { const _Float16* p; int stride; int koff; };

__device__ __forceinline__ v16h frag16(const _Float16* p0, const _Float16* p1) {
  v8h lo = *(const v8h*)p0;     // ds_load_b128
  v8h hi = *(const v8h*)p1;     // ds_load_b128
  v16h r;
#pragma unroll
  for (int i = 0; i < 8; ++i) { r[i] = lo[i]; r[i+8] = hi[i]; }
  return r;
}

// A 16x32 f16 fragment per ISA layout:
// lanes 0-15 (M=0-15): VGPR0-3 = K 0..7, VGPR4-7 = K 16..23; lanes 16-31: +8
__device__ __forceinline__ v16h loadA(const _Float16* A, int stride, int m, int koff, int lh) {
  const _Float16* p = A + m*stride + koff + lh*8;
  return frag16(p, p + 16);
}

// B 32x16 f16 fragment: lane n holds column n; lanes 0-15: K 0..7, K 16..23; lanes 16-31: +8.
// W stored [n][k] in LDS -> two runs of 8 consecutive halfs.
__device__ __forceinline__ v16h loadB(const _Float16* W, int kw, int ncol, int koff, int lh) {
  const _Float16* p = W + ncol*kw + koff + lh*8;
  return frag16(p, p + 16);
}

// Fast sigmoid: v_exp_f32 + v_rcp_f32 (no IEEE division sequence).
__device__ __forceinline__ float fast_sig(float x) {
  return __builtin_amdgcn_rcpf(1.f + __expf(-x));
}
// Fast tanh via tanh(x) = 2*sigmoid(2x) - 1.
__device__ __forceinline__ float fast_tanh(float x) {
  return 2.f * fast_sig(2.f * x) - 1.f;
}

// gates[m][n] = sum_k A[m][k]*W[n][k] + bias[n]; this wave owns gate columns
// [wid*32, wid*32+32) i.e. one LSTM gate. Activation is branch-free:
//   sigmoid gates (i,f,o): sc=1 -> sig(x)
//   cell gate     (g)    : sc=2 -> 2*sig(2x)-1 = tanh(x)
template<int NK>
__device__ __forceinline__ void layer_step(const ASrc* as, const _Float16* W, int kw,
                                           const float* bias, float* gbuf,
                                           int wid, int ln, int lh) {
  const int nt0 = wid*2, nt1 = nt0 + 1;
  const int c0 = nt0*16 + ln, c1 = nt1*16 + ln;
  const float b0v = bias[c0], b1v = bias[c1];
  v8f acc0, acc1;
#pragma unroll
  for (int i = 0; i < 8; ++i) { acc0[i] = b0v; acc1[i] = b1v; }
#pragma unroll
  for (int kb = 0; kb < NK; ++kb) {
    v16h a  = loadA(as[kb].p, as[kb].stride, ln, as[kb].koff, lh);
    v16h b0 = loadB(W, kw, c0, kb*32, lh);
    v16h b1 = loadB(W, kw, c1, kb*32, lh);
    acc0 = __builtin_amdgcn_wmma_f32_16x16x32_f16(false, a, false, b0, (short)0, acc0, false, false);
    acc1 = __builtin_amdgcn_wmma_f32_16x16x32_f16(false, a, false, b1, (short)0, acc1, false, false);
  }
  const float sc  = (wid == 2) ? 2.f : 1.f;   // wave-uniform, branch-free
  const float sub = sc - 1.f;
#pragma unroll
  for (int v = 0; v < 8; ++v) {
    const int row = v + 8*lh;      // C/D layout: VGPR v -> M = v + 8*(lane>=16)
    gbuf[row*GP + c0] = sc * fast_sig(sc * acc0[v]) - sub;
    gbuf[row*GP + c1] = sc * fast_sig(sc * acc1[v]) - sub;
  }
}

// c = f*c + i*g ; h = o*tanh(c). 512 (16x32) slots, cols >= 30 stay zero.
__device__ __forceinline__ void cell_update(const float* gbuf, float* cst, _Float16* hst, int tid) {
#pragma unroll
  for (int q = 0; q < 4; ++q) {
    const int e = q*NTHREADS + tid;        // e == r*32 + c
    const int r = e >> 5, c = e & 31;
    if (c < H_) {
      const float ig = gbuf[r*GP + c];
      const float fg = gbuf[r*GP + 32 + c];
      const float gg = gbuf[r*GP + 64 + c];
      const float og = gbuf[r*GP + 96 + c];
      const float cc = fg * cst[e] + ig * gg;
      cst[e] = cc;
      hst[e] = (_Float16)(og * fast_tanh(cc));
    }
  }
}

} // namespace

__global__ void lstm3_fused_wmma(
    const float* __restrict__ x,
    const float* __restrict__ Wih0, const float* __restrict__ Whh0,
    const float* __restrict__ bih0, const float* __restrict__ bhh0,
    const float* __restrict__ Wih1, const float* __restrict__ Whh1,
    const float* __restrict__ bih1, const float* __restrict__ bhh1,
    const float* __restrict__ Wih2, const float* __restrict__ Whh2,
    const float* __restrict__ bih2, const float* __restrict__ bhh2,
    const float* __restrict__ Wout, const float* __restrict__ bout,
    const float* __restrict__ Wout2, const float* __restrict__ bout2,
    float* __restrict__ out)
{
  extern __shared__ __align__(16) char smem[];
  _Float16* W0    = (_Float16*)(smem + SM_W0);
  _Float16* W1    = (_Float16*)(smem + SM_W1);
  _Float16* W2    = (_Float16*)(smem + SM_W2);
  _Float16* Wo1   = (_Float16*)(smem + SM_WO1);
  _Float16* Wo2   = (_Float16*)(smem + SM_WO2);
  float*    bias0 = (float*)(smem + SM_B0);
  float*    bias1 = (float*)(smem + SM_B1);
  float*    bias2 = (float*)(smem + SM_B2);
  float*    bo1   = (float*)(smem + SM_BO1);
  float*    bo2   = (float*)(smem + SM_BO2);
  _Float16* Ax    = (_Float16*)(smem + SM_AX);
  _Float16* hst   = (_Float16*)(smem + SM_HST);
  float*    cst   = (float*)(smem + SM_CST);
  float*    gbuf  = (float*)(smem + SM_GBUF);
  _Float16* lastb = (_Float16*)(smem + SM_LAST);
  _Float16* obuf  = (_Float16*)(smem + SM_OBUF);
  float*    o2buf = (float*)(smem + SM_O2);

  const int tid  = (int)threadIdx.x;
  const int wid  = tid >> 5;
  const int lane = tid & 31;
  const int ln   = lane & 15;     // WMMA row/col index within tile
  const int lh   = lane >> 4;     // half-wave select
  const int bbase = (int)blockIdx.x * BT;

  // ---- stage weights (padded, [n][k] for contiguous B fragments) ----
  for (int i = tid; i < 128*KW0; i += NTHREADS) {
    const int n = i / KW0, k = i - n*KW0;
    const int gate = n >> 5, j = n & 31;
    float v = 0.f;
    if (j < H_) {
      const int row = gate*H_ + j;
      if (k < IN_)                   v = Wih0[row*IN_ + k];
      else if (k >= 64 && k < 64+H_) v = Whh0[row*H_ + (k - 64)];
    }
    W0[i] = (_Float16)v;
  }
  for (int i = tid; i < 128*KW12; i += NTHREADS) {
    const int n = i / KW12, k = i - n*KW12;
    const int gate = n >> 5, j = n & 31;
    float v1 = 0.f, v2 = 0.f;
    if (j < H_) {
      const int row = gate*H_ + j;
      if (k < H_)                    { v1 = Wih1[row*H_ + k];       v2 = Wih2[row*H_ + k]; }
      else if (k >= 32 && k < 32+H_) { v1 = Whh1[row*H_ + (k-32)];  v2 = Whh2[row*H_ + (k-32)]; }
    }
    W1[i] = (_Float16)v1;
    W2[i] = (_Float16)v2;
  }
  for (int i = tid; i < 32*32; i += NTHREADS) {
    const int n = i >> 5, k = i & 31;
    Wo1[i] = (_Float16)((n < 24 && k < H_) ? Wout[n*H_ + k]  : 0.f);
    Wo2[i] = (_Float16)((n < 24 && k < 24) ? Wout2[n*24 + k] : 0.f);
  }
  for (int i = tid; i < GP; i += NTHREADS) {
    const int gate = i >> 5, j = i & 31;
    bias0[i] = (j < H_) ? bih0[gate*H_ + j] + bhh0[gate*H_ + j] : 0.f;
    bias1[i] = (j < H_) ? bih1[gate*H_ + j] + bhh1[gate*H_ + j] : 0.f;
    bias2[i] = (j < H_) ? bih2[gate*H_ + j] + bhh2[gate*H_ + j] : 0.f;
  }
  if (tid < 32) {
    bo1[tid] = (tid < 24) ? bout[tid]  : 0.f;
    bo2[tid] = (tid < 24) ? bout2[tid] : 0.f;
  }
  for (int i = tid; i < BT*64; i += NTHREADS) Ax[i] = (_Float16)0.f;     // pads stay 0
  for (int i = tid; i < 3*512; i += NTHREADS) { hst[i] = (_Float16)0.f; cst[i] = 0.f; }
  __syncthreads();

  _Float16* h0 = hst;        _Float16* h1 = hst + 512;  _Float16* h2 = hst + 1024;
  float*    c0 = cst;        float*    c1 = cst + 512;  float*    c2 = cst + 1024;

  // ---- fused 3-layer recurrence over time ----
#pragma unroll 1
  for (int t = 0; t < Tt; ++t) {
    // stage x_t tile (f32 -> f16), cols 44..63 remain zero
    for (int i = tid; i < BT*IN_; i += NTHREADS) {
      const int r = i / IN_, k = i - r*IN_;
      Ax[r*64 + k] = (_Float16)x[((size_t)(bbase + r)*Tt + (size_t)t)*IN_ + k];
    }
    if (t + 1 < Tt && tid < BT)  // global_prefetch_b8 for next timestep
      __builtin_prefetch(&x[((size_t)(bbase + tid)*Tt + (size_t)(t+1))*IN_], 0, 1);
    __syncthreads();

    { ASrc as[3] = { {Ax, 64, 0}, {Ax, 64, 32}, {h0, 32, 0} };
      layer_step<3>(as, W0, KW0, bias0, gbuf, wid, ln, lh); }
    __syncthreads();
    cell_update(gbuf, c0, h0, tid);
    __syncthreads();

    { ASrc as[2] = { {h0, 32, 0}, {h1, 32, 0} };
      layer_step<2>(as, W1, KW12, bias1, gbuf, wid, ln, lh); }
    __syncthreads();
    cell_update(gbuf, c1, h1, tid);
    __syncthreads();

    { ASrc as[2] = { {h1, 32, 0}, {h2, 32, 0} };
      layer_step<2>(as, W2, KW12, bias2, gbuf, wid, ln, lh); }
    __syncthreads();
    cell_update(gbuf, c2, h2, tid);
    __syncthreads();
  }

  // ---- head: relu -> (30->24) -> relu -> (24->24) -> softmax ----
  for (int i = tid; i < 512; i += NTHREADS) {
    const float v = (float)h2[i];
    lastb[i] = (_Float16)(v > 0.f ? v : 0.f);
  }
  __syncthreads();

  if (wid < 2) {              // waves 0,1 each own one 16-col N-tile; EXEC all-ones
    const int c = wid*16 + ln;
    const float bv = bo1[c];
    v8f acc;
#pragma unroll
    for (int i = 0; i < 8; ++i) acc[i] = bv;
    v16h a = loadA(lastb, 32, ln, 0, lh);
    v16h b = loadB(Wo1, 32, c, 0, lh);
    acc = __builtin_amdgcn_wmma_f32_16x16x32_f16(false, a, false, b, (short)0, acc, false, false);
#pragma unroll
    for (int v = 0; v < 8; ++v) {
      const int row = v + 8*lh;
      const float r_ = acc[v];
      obuf[row*32 + c] = (_Float16)(r_ > 0.f ? r_ : 0.f);
    }
  }
  __syncthreads();

  if (wid < 2) {
    const int c = wid*16 + ln;
    const float bv = bo2[c];
    v8f acc;
#pragma unroll
    for (int i = 0; i < 8; ++i) acc[i] = bv;
    v16h a = loadA(obuf, 32, ln, 0, lh);
    v16h b = loadB(Wo2, 32, c, 0, lh);
    acc = __builtin_amdgcn_wmma_f32_16x16x32_f16(false, a, false, b, (short)0, acc, false, false);
#pragma unroll
    for (int v = 0; v < 8; ++v) o2buf[(v + 8*lh)*32 + c] = acc[v];
  }
  __syncthreads();

  if (tid < BT) {             // softmax over 24 logits, one row per lane
    const int r = tid;
    float mx = -3.4e38f;
    for (int j = 0; j < 24; ++j) mx = fmaxf(mx, o2buf[r*32 + j]);
    float s = 0.f;
    for (int j = 0; j < 24; ++j) s += __expf(o2buf[r*32 + j] - mx);
    const float inv = 1.f / s;
    for (int j = 0; j < 24; ++j)
      out[(size_t)(bbase + r)*24 + j] = __expf(o2buf[r*32 + j] - mx) * inv;
  }
}

extern "C" void kernel_launch(void* const* d_in, const int* in_sizes, int n_in,
                              void* d_out, int out_size, void* d_ws, size_t ws_size,
                              hipStream_t stream) {
  (void)in_sizes; (void)n_in; (void)out_size; (void)d_ws; (void)ws_size;
  const float* x     = (const float*)d_in[0];
  const float* Wih0  = (const float*)d_in[1];
  const float* Whh0  = (const float*)d_in[2];
  const float* bih0  = (const float*)d_in[3];
  const float* bhh0  = (const float*)d_in[4];
  const float* Wih1  = (const float*)d_in[5];
  const float* Whh1  = (const float*)d_in[6];
  const float* bih1  = (const float*)d_in[7];
  const float* bhh1  = (const float*)d_in[8];
  const float* Wih2  = (const float*)d_in[9];
  const float* Whh2  = (const float*)d_in[10];
  const float* bih2  = (const float*)d_in[11];
  const float* bhh2  = (const float*)d_in[12];
  const float* Wout  = (const float*)d_in[13];
  const float* bout  = (const float*)d_in[14];
  const float* Wout2 = (const float*)d_in[15];
  const float* bout2 = (const float*)d_in[16];
  float* out = (float*)d_out;

  hipFuncSetAttribute((const void*)lstm3_fused_wmma,
                      hipFuncAttributeMaxDynamicSharedMemorySize, SM_TOTAL);
  lstm3_fused_wmma<<<Bsz/BT, NTHREADS, SM_TOTAL, stream>>>(
      x, Wih0, Whh0, bih0, bhh0, Wih1, Whh1, bih1, bhh1,
      Wih2, Whh2, bih2, bhh2, Wout, bout, Wout2, bout2, out);
}
